// GNNEncoder_953482739986
// MI455X (gfx1250) — matrix operand order repaired and checked by reading, not compiled, not run on hardware
//
#include <hip/hip_runtime.h>

// ---------------------------------------------------------------------------
// GAT encoder forward for MI455X (gfx1250), wave32 + WMMA f16->f32.
// All GEMMs use v_wmma_f32_16x16x32_f16. The edge preprocessor is fully fused
// (MLP + LayerNorm + self-loop scatter + per-layer attention dots) so the
// [E,32] edge-feature array never touches HBM. Attention softmax/aggregation
// use atomics into L2-resident node-sized buffers.
// ---------------------------------------------------------------------------

typedef __attribute__((ext_vector_type(16))) _Float16 v16h;
typedef __attribute__((ext_vector_type(8)))  float    v8f;

#define WMMA(a, b, c) __builtin_amdgcn_wmma_f32_16x16x32_f16( \
    false, (a), false, (b), (short)0, (c), false, false)

// Monotone float <-> uint mapping for atomicMax on floats.
__device__ __forceinline__ unsigned fenc(float f) {
  unsigned u = __float_as_uint(f);
  return (u & 0x80000000u) ? ~u : (u | 0x80000000u);
}
__device__ __forceinline__ float fdec(unsigned u) {
  unsigned b = (u & 0x80000000u) ? (u & 0x7fffffffu) : ~u;
  return __uint_as_float(b);
}

// ---------------------------------------------------------------------------
// WMMA GEMM, register-blocked over COLS column tiles:
//   C[M,Nc] = act(A[M,K] @ W[K,Nc] + bias)
// A row-major (f32 or f16), W pre-transposed f16 [Nc][K], C f32 or f16.
// One wave computes a 16 x (16*COLS) output block, reusing the A registers
// across COLS WMMAs per K-step. M%16==0, K%32==0, Nc%(16*COLS)==0.
// ---------------------------------------------------------------------------
template <bool AF16, bool CF16, bool RELU, int COLS>
__global__ __launch_bounds__(32) void gemm16(
    const void* __restrict__ Av, const _Float16* __restrict__ Wt,
    const float* __restrict__ bias, void* __restrict__ Cv, int K, int Nc) {
  const int lane = threadIdx.x;
  const int m  = lane & 15;   // A row
  const int hi = lane >> 4;   // half-wave selector
  const int n  = m;           // B/C column within tile
  const size_t r0 = (size_t)blockIdx.x * 16;
  const int    c0 = blockIdx.y * 16 * COLS;

  v8f acc[COLS];
  #pragma unroll
  for (int j = 0; j < COLS; ++j) acc[j] = (v8f){};

  for (int k0 = 0; k0 < K; k0 += 32) {
    v16h a;
    if (AF16) {
      const _Float16* Ap = (const _Float16*)Av + (r0 + m) * (size_t)K + k0;
      #pragma unroll
      for (int i = 0; i < 8; ++i) {
        a[i]     = Ap[hi * 16 + i];       // K = k0 + hi*16 + i
        a[8 + i] = Ap[hi * 16 + 8 + i];   // K = k0 + hi*16 + 8 + i
      }
    } else {
      const float* Ap = (const float*)Av + (r0 + m) * (size_t)K + k0;
      #pragma unroll
      for (int i = 0; i < 8; ++i) {
        a[i]     = (_Float16)Ap[hi * 16 + i];
        a[8 + i] = (_Float16)Ap[hi * 16 + 8 + i];
      }
    }
    #pragma unroll
    for (int j = 0; j < COLS; ++j) {
      // B: lane n holds a K-contiguous column slice (32-byte aligned)
      v16h b =
          *(const v16h*)(Wt + (size_t)(c0 + j * 16 + n) * K + k0 + hi * 16);
      acc[j] = WMMA(a, b, acc[j]);
    }
  }
  #pragma unroll
  for (int j = 0; j < COLS; ++j) {
    int cn = c0 + j * 16 + n;
    float bv = bias ? bias[cn] : 0.0f;
    #pragma unroll
    for (int r = 0; r < 8; ++r) {
      int mm = r + 8 * hi;  // D layout: VGPR r -> row r / 8+r per half-wave
      float v = acc[j][r] + bv;
      if (RELU) v = fmaxf(v, 0.0f);
      if (CF16)
        ((_Float16*)Cv)[(r0 + mm) * (size_t)Nc + cn] = (_Float16)v;
      else
        ((float*)Cv)[(r0 + mm) * (size_t)Nc + cn] = v;
    }
  }
}

// ---------------------------------------------------------------------------
// Fully fused edge pipeline (one wave per 16 edges, 8 WMMA per tile):
//   t  = LN(ReLU(ea @ W1 + b1) @ W2 + b2)          (kept in LDS only)
//   loopat[dst] += t ; deg[dst] += 1               (self-loop attr scatter)
//   ale[l][edge] = dot(t, w_e[l])  for l = 0..2    (attention edge logits)
// The [E,32] edge feature array never reaches HBM.
// ---------------------------------------------------------------------------
__global__ __launch_bounds__(32) void edge_mlp_scan(
    const float* __restrict__ ea,       // [E,32]
    const _Float16* __restrict__ W1t,   // [64][32] (transposed f16)
    const float* __restrict__ b1,       // [64]
    const _Float16* __restrict__ W2t,   // [32][64]
    const float* __restrict__ b2,       // [32]
    const float* __restrict__ lng, const float* __restrict__ lnb,  // [32]
    const int* __restrict__ dst,        // [E]
    const float* __restrict__ we,       // [3][32] per-layer edge-att weights
    float* __restrict__ deg,            // [N]
    float* __restrict__ loopat,         // [N,32]
    float* __restrict__ ale,            // [3][E+N]
    int EN) {
  __shared__ _Float16 Tsh[16 * 64];
  __shared__ float    T2[16 * 32];
  __shared__ float    mu[16], rinv[16];
  __shared__ int      dsh[16];

  const int lane = threadIdx.x;
  const int m  = lane & 15;
  const int hi = lane >> 4;
  const int n  = m;
  const size_t r0 = (size_t)blockIdx.x * 16;

  // A1: 16x32 tile of edge_attr
  v16h a1;
  const float* Ap = ea + (r0 + m) * 32;
  #pragma unroll
  for (int i = 0; i < 8; ++i) {
    a1[i]     = (_Float16)Ap[hi * 16 + i];
    a1[8 + i] = (_Float16)Ap[hi * 16 + 8 + i];
  }
  // GEMM1: [16,32]@[32,64] -> 4 column tiles; ReLU+bias, stash f16 in LDS
  #pragma unroll
  for (int ct = 0; ct < 4; ++ct) {
    v16h b = *(const v16h*)(W1t + (size_t)(ct * 16 + n) * 32 + hi * 16);
    v8f acc = {};
    acc = WMMA(a1, b, acc);
    float bv = b1[ct * 16 + n];
    #pragma unroll
    for (int r = 0; r < 8; ++r) {
      float v = acc[r] + bv;
      v = v > 0.0f ? v : 0.0f;
      Tsh[(r + 8 * hi) * 64 + ct * 16 + n] = (_Float16)v;
    }
  }
  __syncthreads();
  // GEMM2: [16,64]@[64,32] -> 2 column tiles, K-loop of 2
  #pragma unroll
  for (int ct = 0; ct < 2; ++ct) {
    v8f acc = {};
    #pragma unroll
    for (int ks = 0; ks < 2; ++ks) {
      v16h a2, b;
      const _Float16* Tp = Tsh + m * 64 + ks * 32;
      #pragma unroll
      for (int i = 0; i < 8; ++i) {
        a2[i]     = Tp[hi * 16 + i];
        a2[8 + i] = Tp[hi * 16 + 8 + i];
      }
      b = *(const v16h*)(W2t + (size_t)(ct * 16 + n) * 64 + ks * 32 + hi * 16);
      acc = WMMA(a2, b, acc);
    }
    float bv = b2[ct * 16 + n];
    #pragma unroll
    for (int r = 0; r < 8; ++r)
      T2[(r + 8 * hi) * 32 + ct * 16 + n] = acc[r] + bv;
  }
  __syncthreads();
  // LayerNorm stats per row + edge destinations
  if (lane < 16) {
    float s = 0.f, s2 = 0.f;
    #pragma unroll
    for (int c = 0; c < 32; ++c) {
      float v = T2[lane * 32 + c];
      s += v; s2 += v * v;
    }
    float mean = s * (1.0f / 32.0f);
    float var  = s2 * (1.0f / 32.0f) - mean * mean;
    mu[lane]   = mean;
    rinv[lane] = rsqrtf(var + 1e-5f);
    dsh[lane]  = dst[r0 + lane];
  }
  __syncthreads();
  // Normalize in LDS + scatter into self-loop attr accumulator
  for (int idx = lane; idx < 512; idx += 32) {
    int row = idx >> 5, col = idx & 31;
    float v = (T2[idx] - mu[row]) * rinv[row] * lng[col] + lnb[col];
    T2[idx] = v;
    atomicAdd(loopat + (size_t)dsh[row] * 32 + col, v);
  }
  __syncthreads();
  // Per-layer edge attention logits + degree count
  if (lane < 16) {
    float s0 = 0.f, s1 = 0.f, s2 = 0.f;
    #pragma unroll
    for (int c = 0; c < 32; ++c) {
      float v = T2[lane * 32 + c];
      s0 += v * we[c];
      s1 += v * we[32 + c];
      s2 += v * we[64 + c];
    }
    size_t j = r0 + lane;
    ale[j]                 = s0;
    ale[(size_t)EN + j]    = s1;
    ale[2 * (size_t)EN + j] = s2;
    atomicAdd(deg + dsh[lane], 1.0f);
  }
}

// ---------------------------------------------------------------------------
// Helpers
// ---------------------------------------------------------------------------
__global__ void zero_f32(float* __restrict__ p, int n) {
  int i = blockIdx.x * blockDim.x + threadIdx.x;
  if (i < n) p[i] = 0.0f;
}

// Convert W[K][Nc] f32 row-major -> Wt[Nc][K] f16 (transposed for WMMA B).
__global__ void w_f16t(const float* __restrict__ W, _Float16* __restrict__ Wt,
                       int K, int Nc) {
  int t = blockIdx.x * blockDim.x + threadIdx.x;
  if (t >= K * Nc) return;
  int k = t / Nc, n = t % Nc;
  Wt[(size_t)n * K + k] = (_Float16)W[t];
}

// LayerNorm over 128 channels, one wave32 per row (4 channels/lane).
__global__ __launch_bounds__(256) void ln128(float* __restrict__ X,
                                             const float* __restrict__ g,
                                             const float* __restrict__ b,
                                             int M) {
  int row  = blockIdx.x * 8 + (threadIdx.x >> 5);
  int lane = threadIdx.x & 31;
  if (row >= M) return;
  float4 v = *(float4*)(X + (size_t)row * 128 + lane * 4);
  float s  = v.x + v.y + v.z + v.w;
  float s2 = v.x * v.x + v.y * v.y + v.z * v.z + v.w * v.w;
  #pragma unroll
  for (int o = 16; o > 0; o >>= 1) {
    s  += __shfl_xor(s, o, 32);
    s2 += __shfl_xor(s2, o, 32);
  }
  float mean = s * (1.0f / 128.0f);
  float var  = s2 * (1.0f / 128.0f) - mean * mean;
  float inv  = rsqrtf(var + 1e-5f);
  float4 gg = *(const float4*)(g + lane * 4);
  float4 bb = *(const float4*)(b + lane * 4);
  v.x = (v.x - mean) * inv * gg.x + bb.x;
  v.y = (v.y - mean) * inv * gg.y + bb.y;
  v.z = (v.z - mean) * inv * gg.z + bb.z;
  v.w = (v.w - mean) * inv * gg.w + bb.w;
  *(float4*)(X + (size_t)row * 128 + lane * 4) = v;
}

// Per-channel sum & sum-of-squares over rows (64 channels fixed).
__global__ __launch_bounds__(64) void colsum64(const float* __restrict__ X,
                                               int M, int rows,
                                               float* __restrict__ sums,
                                               float* __restrict__ sumsq) {
  int c = threadIdx.x;
  size_t r0 = (size_t)blockIdx.x * rows;
  float s = 0.f, s2 = 0.f;
  for (int r = 0; r < rows; ++r) {
    size_t rr = r0 + r;
    if (rr >= (size_t)M) break;
    float v = X[rr * 64 + c];
    s += v; s2 += v * v;
  }
  atomicAdd(sums + c, s);
  atomicAdd(sumsq + c, s2);
}

__global__ void bn_finalize(const float* __restrict__ sums,
                            const float* __restrict__ sumsq,
                            const float* __restrict__ g,
                            const float* __restrict__ b,
                            float* __restrict__ scale,
                            float* __restrict__ shift, float invM) {
  int c = threadIdx.x;
  if (c >= 64) return;
  float mean = sums[c] * invM;
  float var  = sumsq[c] * invM - mean * mean;
  float sc   = g[c] * rsqrtf(var + 1e-5f);
  scale[c] = sc;
  shift[c] = b[c] - mean * sc;
}

__global__ void bn_relu(const float* __restrict__ in,
                        const float* __restrict__ sc,
                        const float* __restrict__ sh,
                        float* __restrict__ out, int total) {
  int t = blockIdx.x * blockDim.x + threadIdx.x;
  if (t >= total) return;
  int c = t & 63;
  out[t] = fmaxf(in[t] * sc[c] + sh[c], 0.0f);
}

__global__ void bn_relu_res(const float* __restrict__ agg,
                            const float* __restrict__ sc,
                            const float* __restrict__ sh,
                            float* __restrict__ h, int total) {
  int t = blockIdx.x * blockDim.x + threadIdx.x;
  if (t >= total) return;
  int c = t & 63;
  h[t] += fmaxf(agg[t] * sc[c] + sh[c], 0.0f);
}

// w_e[l][k] = sum_j eW[l][k][j] * a_e[l][j]   (all 3 layers, 96 threads)
__global__ void compute_we(const float* __restrict__ eW,
                           const float* __restrict__ ae,
                           float* __restrict__ we) {
  int t = threadIdx.x;
  if (t >= 96) return;
  int l = t >> 5, k = t & 31;
  float s = 0.f;
  #pragma unroll
  for (int j = 0; j < 64; ++j) s += eW[l * 2048 + k * 64 + j] * ae[l * 64 + j];
  we[t] = s;
}

// Normalize self-loop attrs by degree and emit per-layer self-loop logits.
__global__ void loop_finalize(float* __restrict__ loopat,
                              const float* __restrict__ deg,
                              const float* __restrict__ we,
                              float* __restrict__ ale, int N, int E, int EN) {
  int i = blockIdx.x * blockDim.x + threadIdx.x;
  if (i >= N) return;
  float inv = 1.0f / fmaxf(deg[i], 1.0f);
  float* lp = loopat + (size_t)i * 32;
  float s0 = 0.f, s1 = 0.f, s2 = 0.f;
  #pragma unroll
  for (int c = 0; c < 32; ++c) {
    float v = lp[c] * inv;
    lp[c] = v;
    s0 += v * we[c];
    s1 += v * we[32 + c];
    s2 += v * we[64 + c];
  }
  ale[(size_t)E + i]               = s0;
  ale[(size_t)EN + E + i]          = s1;
  ale[2 * (size_t)EN + E + i]      = s2;
}

// al_s / al_d row dots.
__global__ void rowdots(const float* __restrict__ xs,
                        const float* __restrict__ as,
                        const float* __restrict__ ad,
                        float* __restrict__ al_s, float* __restrict__ al_d,
                        int N) {
  int i = blockIdx.x * blockDim.x + threadIdx.x;
  if (i >= N) return;
  const float* xp = xs + (size_t)i * 64;
  float s = 0.f, d = 0.f;
  #pragma unroll
  for (int c = 0; c < 64; ++c) {
    float v = xp[c];
    s += v * as[c];
    d += v * ad[c];
  }
  al_s[i] = s;
  al_d[i] = d;
}

// Pass 1: attention logits (edges then self-loops) + segment max.
__global__ void attn_logits(const int* __restrict__ src,
                            const int* __restrict__ dst,
                            const float* __restrict__ ale_l,
                            const float* __restrict__ al_s,
                            const float* __restrict__ al_d,
                            float* __restrict__ albuf,
                            unsigned* __restrict__ mxu, int E, int N) {
  int j = blockIdx.x * blockDim.x + threadIdx.x;
  if (j >= E + N) return;
  int s, d;
  if (j < E) { s = src[j]; d = dst[j]; }
  else       { s = d = j - E; }
  float al = al_s[s] + al_d[d] + ale_l[j];
  al = al > 0.0f ? al : 0.2f * al;  // leaky_relu(., 0.2)
  albuf[j] = al;
  atomicMax(mxu + d, fenc(al));
}

// Pass 2: exp(al - max) + segment sum.
__global__ void attn_exp(const int* __restrict__ dst,
                         float* __restrict__ albuf,
                         const unsigned* __restrict__ mxu,
                         float* __restrict__ sm, int E, int N) {
  int j = blockIdx.x * blockDim.x + threadIdx.x;
  if (j >= E + N) return;
  int d = (j < E) ? dst[j] : (j - E);
  float ex = __expf(albuf[j] - fdec(mxu[d]));
  albuf[j] = ex;
  atomicAdd(sm + d, ex);
}

// Pass 3: agg[d] += coef * xs[s]  (thread per edge x 4-channel group).
__global__ void attn_agg(const int* __restrict__ src,
                         const int* __restrict__ dst,
                         const float* __restrict__ albuf,
                         const float* __restrict__ sm,
                         const float* __restrict__ xs,
                         float* __restrict__ agg, int E, int N) {
  int t = blockIdx.x * blockDim.x + threadIdx.x;
  int total = (E + N) * 16;
  if (t >= total) return;
  int j = t >> 4, q = t & 15;
  int s, d;
  if (j < E) { s = src[j]; d = dst[j]; }
  else       { s = d = j - E; }
  float coef = albuf[j] / (sm[d] + 1e-16f);
  float4 v = *(const float4*)(xs + (size_t)s * 64 + q * 4);
  float* o = agg + (size_t)d * 64 + q * 4;
  atomicAdd(o + 0, coef * v.x);
  atomicAdd(o + 1, coef * v.y);
  atomicAdd(o + 2, coef * v.z);
  atomicAdd(o + 3, coef * v.w);
}

// Final: out[c] = relu( mean_rows(h) @ out_W + out_b )
__global__ void final_out(const float* __restrict__ pool,
                          const float* __restrict__ W,
                          const float* __restrict__ b,
                          float* __restrict__ out, float invN) {
  int c = threadIdx.x;
  if (c >= 64) return;
  float s = 0.f;
  #pragma unroll
  for (int k = 0; k < 64; ++k) s += (pool[k] * invN) * W[k * 64 + c];
  out[c] = fmaxf(s + b[c], 0.0f);
}

// ---------------------------------------------------------------------------
// Host orchestration
// ---------------------------------------------------------------------------
static inline int cdiv(int a, int b) { return (a + b - 1) / b; }

extern "C" void kernel_launch(void* const* d_in, const int* in_sizes, int n_in,
                              void* d_out, int out_size, void* d_ws,
                              size_t ws_size, hipStream_t stream) {
  (void)n_in; (void)out_size; (void)ws_size;
  const float* x        = (const float*)d_in[0];
  const int*   ei       = (const int*)  d_in[1];
  const float* ea       = (const float*)d_in[2];
  const float* np_W1    = (const float*)d_in[3];
  const float* np_b1    = (const float*)d_in[4];
  const float* np_W2    = (const float*)d_in[5];
  const float* np_b2    = (const float*)d_in[6];
  const float* np_ln_g  = (const float*)d_in[7];
  const float* np_ln_b  = (const float*)d_in[8];
  const float* ep_W1    = (const float*)d_in[9];
  const float* ep_b1    = (const float*)d_in[10];
  const float* ep_W2    = (const float*)d_in[11];
  const float* ep_b2    = (const float*)d_in[12];
  const float* ep_ln_g  = (const float*)d_in[13];
  const float* ep_ln_b  = (const float*)d_in[14];
  const float* enc_W    = (const float*)d_in[15];
  // d_in[16] enc_b: constant bias before BN -> cancels exactly, unused.
  const float* enc_bn_g = (const float*)d_in[17];
  const float* enc_bn_b = (const float*)d_in[18];
  const float* gat_W    = (const float*)d_in[19];
  const float* gat_eW   = (const float*)d_in[20];
  const float* att_src  = (const float*)d_in[21];
  const float* att_dst  = (const float*)d_in[22];
  const float* att_edge = (const float*)d_in[23];
  // d_in[24] gat_b: constant bias before BN -> cancels exactly, unused.
  const float* bn_g     = (const float*)d_in[25];
  const float* bn_b     = (const float*)d_in[26];
  const float* out_W    = (const float*)d_in[27];
  const float* out_b    = (const float*)d_in[28];

  const int N  = in_sizes[0] / 128;  // 50000 (multiple of 16)
  const int E  = in_sizes[1] / 2;    // 800000 (multiple of 16)
  const int EN = E + N;
  const int* srcI = ei;
  const int* dstI = ei + E;

  // --- workspace bump allocator (256B aligned) ---
  char* wp = (char*)d_ws;
  auto alloc = [&](size_t bytes) -> void* {
    void* r = wp;
    wp += (bytes + 255) & ~(size_t)255;
    return r;
  };
  _Float16* np_W1t = (_Float16*)alloc(128 * 256 * 2);
  _Float16* np_W2t = (_Float16*)alloc(256 * 128 * 2);
  _Float16* ep_W1t = (_Float16*)alloc(32 * 64 * 2);
  _Float16* ep_W2t = (_Float16*)alloc(64 * 32 * 2);
  _Float16* enc_Wt = (_Float16*)alloc(128 * 64 * 2);
  _Float16* gat_Wt = (_Float16*)alloc(3 * 64 * 64 * 2);
  float* we_all = (float*)alloc(96 * 4);
  float* stats  = (float*)alloc(4 * 64 * 4);  // sums|sumsq|scale|shift
  _Float16* t1h = (_Float16*)alloc((size_t)N * 256 * 2);
  float* h0     = (float*)alloc((size_t)N * 128 * 4);
  float* h      = (float*)alloc((size_t)N * 64 * 4);
  float* agg    = (float*)alloc((size_t)N * 64 * 4);
  float* xs     = (float*)alloc((size_t)N * 64 * 4);
  float* al_s   = (float*)alloc((size_t)N * 4);
  float* al_d   = (float*)alloc((size_t)N * 4);
  float* deg    = (float*)alloc((size_t)N * 4);
  float* loopat = (float*)alloc((size_t)N * 32 * 4);
  float* ale    = (float*)alloc((size_t)3 * EN * 4);
  float* albuf  = (float*)alloc((size_t)EN * 4);
  unsigned* mxu = (unsigned*)alloc((size_t)N * 4);
  float* sm     = (float*)alloc((size_t)N * 4);
  float* sums = stats, *sumsq = stats + 64, *scale = stats + 128,
       * shift = stats + 192;

  const int mt = N / 16;  // node row tiles

  // 0) weight transpose/convert to f16 [Nc][K] + per-layer edge-att vectors
  w_f16t<<<cdiv(128 * 256, 256), 256, 0, stream>>>(np_W1, np_W1t, 128, 256);
  w_f16t<<<cdiv(256 * 128, 256), 256, 0, stream>>>(np_W2, np_W2t, 256, 128);
  w_f16t<<<cdiv(32 * 64, 256), 256, 0, stream>>>(ep_W1, ep_W1t, 32, 64);
  w_f16t<<<cdiv(64 * 32, 256), 256, 0, stream>>>(ep_W2, ep_W2t, 64, 32);
  w_f16t<<<cdiv(128 * 64, 256), 256, 0, stream>>>(enc_W, enc_Wt, 128, 64);
  for (int l = 0; l < 3; ++l)
    w_f16t<<<cdiv(64 * 64, 256), 256, 0, stream>>>(gat_W + l * 4096,
                                                   gat_Wt + l * 4096, 64, 64);
  compute_we<<<1, 96, 0, stream>>>(gat_eW, att_edge, we_all);

  // 1) node preprocessor: h0 = LN(relu(x@W1+b1)@W2+b2)
  gemm16<false, true, true, 4><<<dim3(mt, 4), 32, 0, stream>>>(
      x, np_W1t, np_b1, t1h, 128, 256);
  gemm16<true, false, false, 4><<<dim3(mt, 2), 32, 0, stream>>>(
      t1h, np_W2t, np_b2, h0, 256, 128);
  ln128<<<cdiv(N, 8), 256, 0, stream>>>(h0, np_ln_g, np_ln_b, N);

  // 2) fused edge pipeline: MLP+LN in LDS, scatter loopat/deg, edge logits
  zero_f32<<<cdiv(N, 256), 256, 0, stream>>>(deg, N);
  zero_f32<<<cdiv(N * 32, 256), 256, 0, stream>>>(loopat, N * 32);
  edge_mlp_scan<<<E / 16, 32, 0, stream>>>(ea, ep_W1t, ep_b1, ep_W2t, ep_b2,
                                           ep_ln_g, ep_ln_b, dstI, we_all,
                                           deg, loopat, ale, EN);
  loop_finalize<<<cdiv(N, 256), 256, 0, stream>>>(loopat, deg, we_all, ale, N,
                                                  E, EN);

  // 3) node encoder: h = relu(BN(h0 @ enc_W))   (enc_b cancels in BN)
  gemm16<false, false, false, 4><<<dim3(mt, 1), 32, 0, stream>>>(
      h0, enc_Wt, nullptr, agg, 128, 64);
  zero_f32<<<1, 256, 0, stream>>>(stats, 128);
  colsum64<<<cdiv(N, 512), 64, 0, stream>>>(agg, N, 512, sums, sumsq);
  bn_finalize<<<1, 64, 0, stream>>>(sums, sumsq, enc_bn_g, enc_bn_b, scale,
                                    shift, 1.0f / N);
  bn_relu<<<cdiv(N * 64, 256), 256, 0, stream>>>(agg, scale, shift, h, N * 64);

  // 4) GAT layers
  for (int l = 0; l < 3; ++l) {
    gemm16<false, false, false, 4><<<dim3(mt, 1), 32, 0, stream>>>(
        h, gat_Wt + l * 4096, nullptr, xs, 64, 64);
    rowdots<<<cdiv(N, 256), 256, 0, stream>>>(xs, att_src + l * 64,
                                              att_dst + l * 64, al_s, al_d, N);
    zero_f32<<<cdiv(N, 256), 256, 0, stream>>>((float*)mxu, N);  // 0 < enc(-inf)
    zero_f32<<<cdiv(N, 256), 256, 0, stream>>>(sm, N);
    zero_f32<<<cdiv(N * 64, 256), 256, 0, stream>>>(agg, N * 64);
    attn_logits<<<cdiv(EN, 256), 256, 0, stream>>>(
        srcI, dstI, ale + (size_t)l * EN, al_s, al_d, albuf, mxu, E, N);
    attn_exp<<<cdiv(EN, 256), 256, 0, stream>>>(dstI, albuf, mxu, sm, E, N);
    attn_agg<<<cdiv(EN * 16, 256), 256, 0, stream>>>(srcI, dstI, albuf, sm, xs,
                                                     agg, E, N);
    // BN + relu + residual (gat_b cancels in BN)
    zero_f32<<<1, 256, 0, stream>>>(stats, 128);
    colsum64<<<cdiv(N, 512), 64, 0, stream>>>(agg, N, 512, sums, sumsq);
    bn_finalize<<<1, 64, 0, stream>>>(sums, sumsq, bn_g + l * 64, bn_b + l * 64,
                                      scale, shift, 1.0f / N);
    bn_relu_res<<<cdiv(N * 64, 256), 256, 0, stream>>>(agg, scale, shift, h,
                                                       N * 64);
  }

  // 5) global mean pool + output head
  zero_f32<<<1, 256, 0, stream>>>(stats, 128);
  colsum64<<<cdiv(N, 512), 64, 0, stream>>>(h, N, 512, sums, sumsq);
  final_out<<<1, 64, 0, stream>>>(sums, out_W, out_b, (float*)d_out, 1.0f / N);
}
